// GroupedExperts_1992864825973
// MI455X (gfx1250) — compile-verified, matrix operand order checked by reading, not run on hardware
//
#include <hip/hip_runtime.h>

// Grouped experts SwiGLU: h = silu(x@w1) * (x@w3); out = h@w2  (per expert)
// E=8, T=4096, D=2048, H=1024.  fp32 -> bf16 tiles (v_perm packing),
// V_WMMA_F32_16X16X32_BF16 with f32 accumulation, LDS ping-pong double
// buffering (one split-barrier per K-step), ds_load_b128 fragment loads,
// GLOBAL_LOAD_ASYNC_TO_LDS_B128 for the bf16 A-tile in kernel 2.

#define NE 8
#define NT 4096
#define ND 2048
#define NH 1024

typedef __attribute__((ext_vector_type(16))) __bf16 v16bf;
typedef __attribute__((ext_vector_type(8)))  float v8f;

// Scalar fp32 -> bf16, round-to-nearest (ties away): 2 VALU ops.
__device__ __forceinline__ unsigned short f2bf(float f) {
  unsigned int u = __builtin_bit_cast(unsigned int, f);
  return (unsigned short)((u + 0x8000u) >> 16);
}

// Pack two fp32 into a packed-bf16 dword (lo -> [15:0], hi -> [31:16]).
// Fallback: two rounding adds + one v_perm_b32 byte-select (bytes 7,6 of hi
// and 3,2 of lo -> selector 0x07060302), ~1.5 VALU per element.
__device__ __forceinline__ unsigned int pk_bf16(float lo, float hi) {
#if __has_builtin(__builtin_amdgcn_cvt_pk_bf16_f32)
  auto r = __builtin_amdgcn_cvt_pk_bf16_f32(lo, hi);
  static_assert(sizeof(r) == 4, "cvt_pk_bf16_f32 result must be 32-bit");
  return __builtin_bit_cast(unsigned int, r);
#else
  unsigned int ulo = __builtin_bit_cast(unsigned int, lo) + 0x8000u;
  unsigned int uhi = __builtin_bit_cast(unsigned int, hi) + 0x8000u;
  return __builtin_amdgcn_perm(uhi, ulo, 0x07060302u);
#endif
}

// Load one 16x32 bf16 WMMA fragment for this lane from an LDS tile stored with
// 32 contiguous bf16 per row (row = M for A, row = N for the B^T tile).
// 16-bit operand layout (ISA 7.12.2): lanes 0-15 take K {0..7, 16..23},
// lanes 16-31 take K {8..15, 24..31}  -> two ds_load_b128 per fragment.
__device__ __forceinline__ v16bf frag_ld(const unsigned short* rowbase, int khalf) {
  union { v16bf v; uint4 q[2]; } f;
  f.q[0] = *(const uint4*)(rowbase + khalf * 8);
  f.q[1] = *(const uint4*)(rowbase + 16 + khalf * 8);
  return f.v;
}

__device__ __forceinline__ v8f wmma_bf16(v16bf a, v16bf b, v8f c) {
  return __builtin_amdgcn_wmma_f32_16x16x32_bf16(false, a, false, b, (short)0, c,
                                                 false, false);
}

// Async copy 16B/lane global -> LDS (GV mode, ASYNCcnt-tracked).
__device__ __forceinline__ void async_g2l_b128(const void* gaddr, void* lds) {
  unsigned ldsoff = (unsigned)(unsigned long long)lds;  // LDS aperture: low 32 bits
  asm volatile("global_load_async_to_lds_b128 %0, %1, off"
               :: "v"(ldsoff), "v"(gaddr) : "memory");
}
__device__ __forceinline__ void wait_async0() {
  asm volatile("s_wait_asynccnt 0x0" ::: "memory");
}

// ---------------------------------------------------------------------------
// Kernel 1: C1 = X@W1, C3 = X@W3 (128x64 tile / block), h = silu(C1)*C3 -> bf16
// ---------------------------------------------------------------------------
__global__ __launch_bounds__(256) void moe_gemm13_silu(
    const float* __restrict__ x, const float* __restrict__ w1,
    const float* __restrict__ w3, unsigned short* __restrict__ hbuf) {
  __shared__ __align__(16) unsigned short lA [2][128 * 32];
  __shared__ __align__(16) unsigned short lB1[2][ 64 * 32];
  __shared__ __align__(16) unsigned short lB3[2][ 64 * 32];

  const int tid   = threadIdx.x;
  const int lane  = tid & 31;
  const int wave  = tid >> 5;
  const int waveM = wave >> 1;     // 0..3  -> 32-row slab
  const int waveN = wave & 1;      // 0..1  -> 32-col slab
  const int lid   = lane & 15;
  const int khalf = lane >> 4;

  const int e       = blockIdx.z;
  const int rowBase = blockIdx.y * 128;
  const int nBase   = blockIdx.x * 64;

  const float* xe  = x  + (size_t)e * NT * ND + (size_t)rowBase * ND;
  const float* w1e = w1 + (size_t)e * ND * NH + nBase;
  const float* w3e = w3 + (size_t)e * ND * NH + nBase;

  const v8f zero = {0.f, 0.f, 0.f, 0.f, 0.f, 0.f, 0.f, 0.f};
  v8f c1[2][2], c3[2][2];
#pragma unroll
  for (int i = 0; i < 2; ++i)
#pragma unroll
    for (int j = 0; j < 2; ++j) { c1[i][j] = zero; c3[i][j] = zero; }

  // A-tile staging: one float4 (4 K) per thread per pass, 32 rows/pass
  const int a_kv = tid & 7;
  const int a_r0 = tid >> 3;
  // B-tile staging: fixed N per thread, 4 K-consecutive elems per pass
  const int b_n  = tid & 63;
  const int b_k4 = (tid >> 6) * 4;   // {0,4,8,12}; +16 on second pass

  auto stage = [&](int k0, int buf) {
#pragma unroll
    for (int p = 0; p < 4; ++p) {
      const int r = a_r0 + p * 32;
      const float* gp = xe + (size_t)r * ND + k0 + a_kv * 4;
      if (k0 + 32 < ND) __builtin_prefetch(gp + 32, 0, 1);
      float4 v = *(const float4*)gp;
      uint2 pk;
      pk.x = pk_bf16(v.x, v.y);
      pk.y = pk_bf16(v.z, v.w);
      *(uint2*)&lA[buf][r * 32 + a_kv * 4] = pk;
    }
#pragma unroll
    for (int p = 0; p < 2; ++p) {
      const int k = b_k4 + p * 16;
      const float* g1 = w1e + (size_t)(k0 + k) * NH + b_n;
      const float* g3 = w3e + (size_t)(k0 + k) * NH + b_n;
      if (k0 + 32 < ND) __builtin_prefetch(g1 + (size_t)32 * NH, 0, 1);
      uint2 p1, p3;
      p1.x = pk_bf16(g1[0],            g1[(size_t)NH]);
      p1.y = pk_bf16(g1[(size_t)2*NH], g1[(size_t)3*NH]);
      p3.x = pk_bf16(g3[0],            g3[(size_t)NH]);
      p3.y = pk_bf16(g3[(size_t)2*NH], g3[(size_t)3*NH]);
      *(uint2*)&lB1[buf][b_n * 32 + k] = p1;
      *(uint2*)&lB3[buf][b_n * 32 + k] = p3;
    }
  };

  stage(0, 0);
  __syncthreads();

  const int steps = ND / 32;
  for (int kt = 0; kt < steps; ++kt) {
    const int buf = kt & 1;
    if (kt + 1 < steps) stage((kt + 1) * 32, buf ^ 1);

    v16bf a0  = frag_ld(&lA [buf][(waveM * 32 +      lid) * 32], khalf);
    v16bf a1  = frag_ld(&lA [buf][(waveM * 32 + 16 + lid) * 32], khalf);
    v16bf b10 = frag_ld(&lB1[buf][(waveN * 32 +      lid) * 32], khalf);
    v16bf b11 = frag_ld(&lB1[buf][(waveN * 32 + 16 + lid) * 32], khalf);
    v16bf b30 = frag_ld(&lB3[buf][(waveN * 32 +      lid) * 32], khalf);
    v16bf b31 = frag_ld(&lB3[buf][(waveN * 32 + 16 + lid) * 32], khalf);

    c1[0][0] = wmma_bf16(a0, b10, c1[0][0]);
    c1[0][1] = wmma_bf16(a0, b11, c1[0][1]);
    c1[1][0] = wmma_bf16(a1, b10, c1[1][0]);
    c1[1][1] = wmma_bf16(a1, b11, c1[1][1]);
    c3[0][0] = wmma_bf16(a0, b30, c3[0][0]);
    c3[0][1] = wmma_bf16(a0, b31, c3[0][1]);
    c3[1][0] = wmma_bf16(a1, b30, c3[1][0]);
    c3[1][1] = wmma_bf16(a1, b31, c3[1][1]);

    __syncthreads();
  }

  // Epilogue: h = silu(c1) * c3 -> bf16.
  // C/D layout: VGPR r, lane l -> M = r + 8*(l>>4), N = l&15.
  unsigned short* hout = hbuf + (size_t)e * NT * NH;
#pragma unroll
  for (int mi = 0; mi < 2; ++mi)
#pragma unroll
    for (int ni = 0; ni < 2; ++ni) {
      const int col = nBase + waveN * 32 + ni * 16 + lid;
#pragma unroll
      for (int r = 0; r < 8; ++r) {
        const int row = rowBase + waveM * 32 + mi * 16 + r + 8 * khalf;
        const float g = c1[mi][ni][r];
        const float s = g / (1.f + __expf(-g));
        hout[(size_t)row * NH + col] = f2bf(s * c3[mi][ni][r]);
      }
    }
}

// ---------------------------------------------------------------------------
// Kernel 2: out = h(bf16) @ W2 -> fp32 (128x128 tile / block, 8 waves @ 32x64)
// ---------------------------------------------------------------------------
__global__ __launch_bounds__(256) void moe_gemm2(
    const unsigned short* __restrict__ hbuf, const float* __restrict__ w2,
    float* __restrict__ out) {
  __shared__ __align__(16) unsigned short lA[2][128 * 32];
  __shared__ __align__(16) unsigned short lB[2][128 * 32];

  const int tid   = threadIdx.x;
  const int lane  = tid & 31;
  const int wave  = tid >> 5;
  const int waveM = wave >> 1;     // 0..3 -> 32-row slab
  const int waveN = wave & 1;      // 0..1 -> 64-col slab
  const int lid   = lane & 15;
  const int khalf = lane >> 4;

  const int e       = blockIdx.z;
  const int rowBase = blockIdx.y * 128;
  const int nBase   = blockIdx.x * 128;

  const unsigned short* he = hbuf + (size_t)e * NT * NH + (size_t)rowBase * NH;
  const float*         w2e = w2   + (size_t)e * NH * ND + nBase;

  const v8f zero = {0.f, 0.f, 0.f, 0.f, 0.f, 0.f, 0.f, 0.f};
  v8f acc[2][4];
#pragma unroll
  for (int i = 0; i < 2; ++i)
#pragma unroll
    for (int j = 0; j < 4; ++j) acc[i][j] = zero;

  const int a_qv = tid & 3;          // which uint4 (8 bf16) along K
  const int a_r0 = tid >> 2;         // 64 rows per pass
  const int b_n  = tid & 127;        // fixed N per thread
  const int b_k4 = (tid >> 7) * 4;   // {0,4}; +8 per pass

  auto stage = [&](int k0, int buf) {
    // A tile (already bf16): async global -> LDS, 16B per lane per pass.
#pragma unroll
    for (int p = 0; p < 2; ++p) {
      const int r = a_r0 + p * 64;
      async_g2l_b128(he + (size_t)r * NH + k0 + a_qv * 8,
                     &lA[buf][r * 32 + a_qv * 8]);
    }
    // B tile: 32 K x 128 N fp32, convert + transpose to [n][k] bf16.
#pragma unroll
    for (int p = 0; p < 4; ++p) {
      const int k = b_k4 + p * 8;
      const float* gp = w2e + (size_t)(k0 + k) * ND + b_n;
      if (k0 + 32 < NH) __builtin_prefetch(gp + (size_t)32 * ND, 0, 1);
      uint2 pk;
      pk.x = pk_bf16(gp[0],            gp[(size_t)ND]);
      pk.y = pk_bf16(gp[(size_t)2*ND], gp[(size_t)3*ND]);
      *(uint2*)&lB[buf][b_n * 32 + k] = pk;
    }
  };

  stage(0, 0);
  wait_async0();
  __syncthreads();

  const int steps = NH / 32;
  for (int kt = 0; kt < steps; ++kt) {
    const int buf = kt & 1;
    if (kt + 1 < steps) stage((kt + 1) * 32, buf ^ 1);

    v16bf a0 = frag_ld(&lA[buf][(waveM * 32 +      lid) * 32], khalf);
    v16bf a1 = frag_ld(&lA[buf][(waveM * 32 + 16 + lid) * 32], khalf);
    v16bf b[4];
#pragma unroll
    for (int j = 0; j < 4; ++j)
      b[j] = frag_ld(&lB[buf][(waveN * 64 + j * 16 + lid) * 32], khalf);

#pragma unroll
    for (int j = 0; j < 4; ++j) {
      acc[0][j] = wmma_bf16(a0, b[j], acc[0][j]);
      acc[1][j] = wmma_bf16(a1, b[j], acc[1][j]);
    }

    wait_async0();   // ping buffer's async A-tile writes complete
    __syncthreads(); // all waves done reading current buffer
  }

  float* oute = out + (size_t)e * NT * ND;
#pragma unroll
  for (int mi = 0; mi < 2; ++mi)
#pragma unroll
    for (int ni = 0; ni < 4; ++ni) {
      const int col = nBase + waveN * 64 + ni * 16 + lid;
#pragma unroll
      for (int r = 0; r < 8; ++r) {
        const int row = rowBase + waveM * 32 + mi * 16 + r + 8 * khalf;
        oute[(size_t)row * ND + col] = acc[mi][ni][r];
      }
    }
}

extern "C" void kernel_launch(void* const* d_in, const int* in_sizes, int n_in,
                              void* d_out, int out_size, void* d_ws, size_t ws_size,
                              hipStream_t stream) {
  const float* x  = (const float*)d_in[0];
  const float* w1 = (const float*)d_in[1];
  const float* w2 = (const float*)d_in[2];
  const float* w3 = (const float*)d_in[3];
  float* out = (float*)d_out;
  unsigned short* hbuf = (unsigned short*)d_ws;  // E*T*H bf16 = 64 MB

  dim3 blk(256, 1, 1);
  dim3 g1(NH / 64,  NT / 128, NE);  // 16 x 32 x 8
  dim3 g2(ND / 128, NT / 128, NE);  // 16 x 32 x 8
  hipLaunchKernelGGL(moe_gemm13_silu, g1, blk, 0, stream, x, w1, w3, hbuf);
  hipLaunchKernelGGL(moe_gemm2,       g2, blk, 0, stream, hbuf, w2, out);
}
